// VisualInputEmbedding_dfs_58643483459633
// MI455X (gfx1250) — matrix operand order repaired and checked by reading, not compiled, not run on hardware
//
#include <hip/hip_runtime.h>
#include <hip/hip_bf16.h>

// ---------------------------------------------------------------------------
// Types (gfx1250, wave32)
// ---------------------------------------------------------------------------
typedef __attribute__((ext_vector_type(16))) __bf16 v16bf;
typedef __attribute__((ext_vector_type(8)))  __bf16 v8bf;
typedef __attribute__((ext_vector_type(8)))  float  v8f;
typedef __attribute__((ext_vector_type(4)))  float  f32x4;
typedef __attribute__((ext_vector_type(4)))  unsigned int v4u;
typedef __attribute__((ext_vector_type(8)))  int    v8i;
typedef __attribute__((ext_vector_type(4)))  int    v4i;

#define HID   768
#define TTOK  344
#define BATCH 64
#define NROWS (BATCH * TTOK)                    // 22016 token rows
#define MASK_OFF ((size_t)BATCH * TTOK * HID)   // 16908288
#define TOK_BYTES ((size_t)NROWS * HID * 4)     // tokens scratch: 67.6 MB

// GEMM tiling: block 128(M) x 256(N), K-step 64; 8 waves, each wave 64x64
#define BM 128
#define BN 256
#define BK 64
#define AST (BK + 8)          // LDS A row stride in halves (144B)
#define BST (BK + 8)          // LDS B row stride in halves (144B, from TDM padding)
#define A_BYTES   (BM * AST * 2)            // 18432
#define BBUF_BYTES (BN * BST * 2)           // 36864
#define SMEM_BYTES (A_BYTES + 2 * BBUF_BYTES)  // 92160 (dynamic LDS)

__device__ __forceinline__ v16bf combine16(v8bf lo, v8bf hi) {
    v16bf r;
#pragma unroll
    for (int i = 0; i < 8; ++i) { r[i] = lo[i]; r[i + 8] = hi[i]; }
    return r;
}

// ---------------------------------------------------------------------------
// W pre-pass: [K, HID] f32  ->  [HID, K] bf16 (transposed, WMMA-B friendly)
// ---------------------------------------------------------------------------
__global__ __launch_bounds__(256)
void vie_wt_prep(const float* __restrict__ W, unsigned short* __restrict__ Wt_, int K) {
    __shared__ __bf16 tile[32][33];
    __bf16* Wt = (__bf16*)Wt_;
    const int hb = blockIdx.x * 32;   // column block in W (= row block in Wt)
    const int kb = blockIdx.y * 32;
    const int tx = threadIdx.x & 31;
    const int ty = threadIdx.x >> 5;  // 0..7
#pragma unroll
    for (int i = 0; i < 32; i += 8)
        tile[ty + i][tx] = (__bf16)W[(size_t)(kb + ty + i) * HID + hb + tx];
    __syncthreads();
#pragma unroll
    for (int i = 0; i < 32; i += 8)
        Wt[(size_t)(hb + ty + i) * K + kb + tx] = tile[tx][ty + i];
}

// ---------------------------------------------------------------------------
// Tiled bf16 WMMA GEMM.
//   A: X [M,K] f32, converted to bf16 through registers into LDS.
//   B: Wt [HID,K] bf16, staged by the Tensor Data Mover (double-buffered LDS).
//   Y: tokens [BATCH, TTOK, HID] f32; rows remapped for the modality concat.
// ---------------------------------------------------------------------------
__global__ __launch_bounds__(256)
void vie_wmma_gemm(const float* __restrict__ X, const unsigned short* __restrict__ Wt_,
                   const float* __restrict__ bias, float* __restrict__ Y,
                   int K, int segT, int segOff) {
    extern __shared__ char smem_raw[];
    __bf16* Alds = (__bf16*)smem_raw;
    const __bf16* Wt = (const __bf16*)Wt_;

    const int tid  = threadIdx.x;
    const int lane = tid & 31;
    const int wid  = tid >> 5;
    const int wm   = wid & 1;      // 2 waves across M (64 rows each)
    const int wn   = wid >> 1;     // 4 waves across N (64 cols each)
    const int m0   = blockIdx.y * BM;
    const int n0   = blockIdx.x * BN;

    v8f acc[4][4];
    const v8f vzero = {0.f, 0.f, 0.f, 0.f, 0.f, 0.f, 0.f, 0.f};
#pragma unroll
    for (int mt = 0; mt < 4; ++mt)
#pragma unroll
        for (int nt = 0; nt < 4; ++nt) acc[mt][nt] = vzero;

    // ---- TDM: async 2D tile load of B (bf16 [256 n][64 k]) into LDS, padded
    //      so each 128B row gets +16B -> 144B row stride (= BST halves).
    auto tdm_issue = [&](int kt, int buf) {
        unsigned lds_addr = (unsigned)(size_t)smem_raw + (unsigned)A_BYTES
                          + (unsigned)buf * (unsigned)BBUF_BYTES;   // low32 of generic = LDS offset
        unsigned long long ga =
            (unsigned long long)(size_t)(Wt + (size_t)n0 * K + (size_t)kt * BK);
        v4u g0;
        g0[0] = 1u;                                        // count=1 (valid user D#)
        g0[1] = lds_addr;                                  // LDS byte address
        g0[2] = (unsigned)(ga & 0xffffffffu);              // global_addr[31:0]
        g0[3] = (unsigned)((ga >> 32) & 0x01ffffffu) | (2u << 30);  // addr[56:32] | type=2
        v8i g1;
        g1[0] = (int)((1u << 16)      // data_size = 2 bytes
                    | (1u << 20)      // pad_enable
                    | (4u << 22)      // pad_interval: 32 DWORDs (=128B)
                    | (3u << 25));    // pad_amount:   4 DWORDs (=16B)
        g1[1] = (int)(((unsigned)K & 0xffffu) << 16);      // tensor_dim0 lo16 (K halves)
        g1[2] = (int)(((unsigned)K >> 16) | ((unsigned)HID << 16)); // dim0 hi | dim1 lo
        g1[3] = (int)((unsigned)BK << 16);                 // dim1 hi=0 | tile_dim0=64
        g1[4] = (int)(unsigned)BN;                         // tile_dim1=256 | tile_dim2=0
        g1[5] = (int)(unsigned)K;                          // tensor_dim0_stride lo32 (halves)
        g1[6] = 0;                                         // stride0 hi | stride1 lo
        g1[7] = 0;                                         // stride1 hi
        v4i z4 = {0, 0, 0, 0};
        v8i z8 = {0, 0, 0, 0, 0, 0, 0, 0};
        __builtin_amdgcn_tensor_load_to_lds(g0, g1, z4, z4, z8, 0);
    };

    // ---- A prefetch registers (f32 tile staged in regs, converted on LDS store)
    f32x4 pa[4][2];
    auto loadA = [&](int kt) {
#pragma unroll
        for (int j = 0; j < 4; ++j) {
            int i  = tid + 256 * j;
            int r  = i >> 3;             // 0..127
            int c8 = (i & 7) * 8;        // 0..56
            const float* p = X + (size_t)(m0 + r) * K + kt * BK + c8;
            pa[j][0] = *(const f32x4*)(p);
            pa[j][1] = *(const f32x4*)(p + 4);
        }
    };
    auto storeA = [&]() {
#pragma unroll
        for (int j = 0; j < 4; ++j) {
            int i  = tid + 256 * j;
            int r  = i >> 3;
            int c8 = (i & 7) * 8;
            v8bf p;
#pragma unroll
            for (int e = 0; e < 4; ++e) { p[e] = (__bf16)pa[j][0][e]; p[e + 4] = (__bf16)pa[j][1][e]; }
            *(v8bf*)&Alds[r * AST + c8] = p;
        }
    };

    const int nKT = K / BK;
    if (wid == 0) tdm_issue(0, 0);     // kick off B tile 0 DMA
    loadA(0);
    for (int kt = 0; kt < nKT; ++kt) {
        if (kt) __syncthreads();                      // prev compute done (A + B alt buf free)
        storeA();
        if (wid == 0) __builtin_amdgcn_s_wait_tensorcnt(0);   // B tile kt in LDS
        __syncthreads();                              // publish A, broadcast B arrival
        if (kt + 1 < nKT) {
            if (wid == 0) tdm_issue(kt + 1, (kt + 1) & 1);    // DMA overlaps WMMAs
            loadA(kt + 1);                                    // global latency overlaps WMMAs
        }

        const __bf16* Bb = (const __bf16*)(smem_raw + A_BYTES + (kt & 1) * BBUF_BYTES);
#pragma unroll
        for (int ks = 0; ks < BK / 32; ++ks) {
            const int aoff = ks * 32 + ((lane & 16) ? 8 : 0);
            const int boff = ks * 32 + ((lane & 16) ? 8 : 0);
            v16bf a[4], b[4];
#pragma unroll
            for (int mt = 0; mt < 4; ++mt) {
                const __bf16* ap = &Alds[(wm * 64 + mt * 16 + (lane & 15)) * AST + aoff];
                a[mt] = combine16(*(const v8bf*)ap, *(const v8bf*)(ap + 16));
            }
#pragma unroll
            for (int nt = 0; nt < 4; ++nt) {
                const __bf16* bp = &Bb[(wn * 64 + nt * 16 + (lane & 15)) * BST + boff];
                b[nt] = combine16(*(const v8bf*)bp, *(const v8bf*)(bp + 16));
            }
#pragma unroll
            for (int mt = 0; mt < 4; ++mt)
#pragma unroll
                for (int nt = 0; nt < 4; ++nt)
                    acc[mt][nt] = __builtin_amdgcn_wmma_f32_16x16x32_bf16(
                        false, a[mt], false, b[nt], (short)0, acc[mt][nt],
                        false, false);
        }
    }

    // ---- epilogue: add bias, scatter rows into concat [BATCH, TTOK, HID]
    int   cols[4];
    float bvs[4];
#pragma unroll
    for (int nt = 0; nt < 4; ++nt) {
        cols[nt] = n0 + wn * 64 + nt * 16 + (lane & 15);
        bvs[nt]  = bias[cols[nt]];
    }
#pragma unroll
    for (int mt = 0; mt < 4; ++mt) {
        const int g0 = m0 + wm * 64 + mt * 16 + ((lane & 16) ? 8 : 0);
        const unsigned q = (unsigned)g0 / (unsigned)segT;   // one div per m-tile
        const int rem  = g0 - (int)q * segT;
        const int base = (int)q * TTOK + segOff;
#pragma unroll
        for (int r = 0; r < 8; ++r) {
            int t    = rem + r;                               // wraps at most once
            int orow = base + t + ((t >= segT) ? (TTOK - segT) : 0);
            size_t off = (size_t)orow * HID;
#pragma unroll
            for (int nt = 0; nt < 4; ++nt)
                Y[off + cols[nt]] = acc[mt][nt][r] + bvs[nt];
        }
    }
}

// ---------------------------------------------------------------------------
// Per-token-row epilogue: x += order_emb[order_ids]; LayerNorm; out + mask.
// One wave32 per row (24 columns per lane), 8 rows per 256-thread block.
// ---------------------------------------------------------------------------
__global__ __launch_bounds__(256)
void vie_ln_epilogue(const float* __restrict__ tokens, const int* __restrict__ ids,
                     const float* __restrict__ emb, const float* __restrict__ gamma,
                     const float* __restrict__ beta, float* __restrict__ out) {
    const int wid  = threadIdx.x >> 5;
    const int lane = threadIdx.x & 31;
    const int row  = blockIdx.x * 8 + wid;      // 2752 * 8 == 22016 exactly

    const float* xr = tokens + (size_t)row * HID;
    const int id    = ids[row];
    const float* er = emb + (size_t)id * HID;

    float x[24];
    float s = 0.f, s2 = 0.f;
#pragma unroll
    for (int j = 0; j < 24; ++j) {
        int c = lane + 32 * j;
        float v = xr[c] + er[c];
        x[j] = v;
        s  += v;
        s2 += v * v;
    }
#pragma unroll
    for (int o = 16; o > 0; o >>= 1) {          // wave32 butterfly reduction
        s  += __shfl_xor(s,  o, 32);
        s2 += __shfl_xor(s2, o, 32);
    }
    const float inv  = 1.0f / (float)HID;
    const float mu   = s * inv;
    const float var  = s2 * inv - mu * mu;
    const float rstd = rsqrtf(var + 1e-12f);

    float* orw = out + (size_t)row * HID;
#pragma unroll
    for (int j = 0; j < 24; ++j) {
        int c = lane + 32 * j;
        orw[c] = (x[j] - mu) * rstd * gamma[c] + beta[c];
    }
    if (lane == 0) out[MASK_OFF + row] = 1.0f;  // non_pad_mask (all true)
}

// ---------------------------------------------------------------------------
// Host launch
// ---------------------------------------------------------------------------
extern "C" void kernel_launch(void* const* d_in, const int* in_sizes, int n_in,
                              void* d_out, int out_size, void* d_ws, size_t ws_size,
                              hipStream_t stream) {
    (void)in_sizes; (void)n_in; (void)out_size; (void)ws_size;

    const float* obj_feats    = (const float*)d_in[0];   // [64,160,2048]
    const float* rel_feats    = (const float*)d_in[1];   // [64,160,1024]
    const float* frame_feats  = (const float*)d_in[2];   // [64,16,2048]
    const float* action_feats = (const float*)d_in[3];   // [64,8,2048]
    const int*   order_ids    = (const int*)  d_in[4];   // [64,344]
    const float* W_obj    = (const float*)d_in[5];
    const float* b_obj    = (const float*)d_in[6];
    const float* W_rel    = (const float*)d_in[7];
    const float* b_rel    = (const float*)d_in[8];
    const float* W_frame  = (const float*)d_in[9];
    const float* b_frame  = (const float*)d_in[10];
    const float* W_action = (const float*)d_in[11];
    const float* b_action = (const float*)d_in[12];
    const float* order_emb = (const float*)d_in[13];     // [512,768]
    const float* ln_gamma  = (const float*)d_in[14];
    const float* ln_beta   = (const float*)d_in[15];

    float* tokens = (float*)d_ws;                        // [64,344,768] f32
    float* out    = (float*)d_out;

    // bf16 transposed weights live after the tokens buffer in d_ws
    unsigned short* wt_base   = (unsigned short*)((char*)d_ws + TOK_BYTES);
    unsigned short* Wt_obj    = wt_base;                                   // 768x2048
    unsigned short* Wt_rel    = wt_base + (size_t)HID * 2048;              // 768x1024
    unsigned short* Wt_frame  = wt_base + (size_t)HID * (2048 + 1024);     // 768x2048
    unsigned short* Wt_action = wt_base + (size_t)HID * (2048 + 1024 + 2048);

    const dim3 blk(256);
    // one-time W transpose/convert to bf16 (11 MB total)
    vie_wt_prep<<<dim3(HID / 32, 2048 / 32), blk, 0, stream>>>(W_obj,    Wt_obj,    2048);
    vie_wt_prep<<<dim3(HID / 32, 1024 / 32), blk, 0, stream>>>(W_rel,    Wt_rel,    1024);
    vie_wt_prep<<<dim3(HID / 32, 2048 / 32), blk, 0, stream>>>(W_frame,  Wt_frame,  2048);
    vie_wt_prep<<<dim3(HID / 32, 2048 / 32), blk, 0, stream>>>(W_action, Wt_action, 2048);

    // segment GEMMs: grid = (HID/BN, M/BM), dynamic LDS for A + double-buffered B
    vie_wmma_gemm<<<dim3(HID / BN, (BATCH * 160) / BM), blk, SMEM_BYTES, stream>>>(
        obj_feats, Wt_obj, b_obj, tokens, 2048, 160, 0);
    vie_wmma_gemm<<<dim3(HID / BN, (BATCH * 160) / BM), blk, SMEM_BYTES, stream>>>(
        rel_feats, Wt_rel, b_rel, tokens, 1024, 160, 160);
    vie_wmma_gemm<<<dim3(HID / BN, (BATCH * 16) / BM), blk, SMEM_BYTES, stream>>>(
        frame_feats, Wt_frame, b_frame, tokens, 2048, 16, 320);
    vie_wmma_gemm<<<dim3(HID / BN, (BATCH * 8) / BM), blk, SMEM_BYTES, stream>>>(
        action_feats, Wt_action, b_action, tokens, 2048, 8, 336);

    // gather + layernorm + mask: one wave per token row
    vie_ln_epilogue<<<dim3(NROWS / 8), blk, 0, stream>>>(
        tokens, order_ids, order_emb, ln_gamma, ln_beta, out);
}
